// InteractionLayer_11390253269007
// MI455X (gfx1250) — matrix operand order repaired and checked by reading, not compiled, not run on hardware
//
#include <hip/hip_runtime.h>
#include <hip/hip_bf16.h>

// ---------------------------------------------------------------------------
// InteractionLayer for MI455X (gfx1250, wave32, WMMA)
// B=32, NV=512, NL=256, D=1024, H=4096
// fp32 -> bf16; all GEMMs NT-form v_wmma_f32_16x16x32_bf16, fp32 accumulate.
// Block tile 128x256x32, 8 wave32s (2x4), wave tile 64x64 -> 16 wmma / K-step.
// Double-buffered LDS staging via global_load_async_to_lds_b128 when available.
// ---------------------------------------------------------------------------

typedef __attribute__((ext_vector_type(16))) __bf16 v16bf;
typedef __attribute__((ext_vector_type(8)))  __bf16 v8bf;
typedef __attribute__((ext_vector_type(4)))  __bf16 v4bf;
typedef __attribute__((ext_vector_type(8)))  float  v8f;

static constexpr int Bb_  = 32;
static constexpr int NV_  = 512;
static constexpr int NL_  = 256;
static constexpr int D_   = 1024;
static constexpr int H_   = 4096;
static constexpr int KD_  = NL_ + NV_;        // 768
static constexpr int MV_  = Bb_ * NV_;        // 16384 tokens of V
static constexpr int ML_  = Bb_ * NL_;        // 8192 tokens of L

// ---------------------------------------------------------------------------
// CDNA5 async global->LDS copy (ASYNCcnt-tracked, bypasses VGPRs).
// Builtin signature (from compiler diagnostic): first arg is a pointer to
// int __vector(4) in the global (__device__/AS1) address space; LDS arg in AS3.
// Falls back to a sync load/store path if builtins are absent.
// ---------------------------------------------------------------------------
#if defined(__gfx1250__) && __has_builtin(__builtin_amdgcn_global_load_async_to_lds_b128)
#define USE_ASYNC_LDS 1
typedef int v4i_vs __attribute__((vector_size(16)));
typedef __attribute__((address_space(1))) v4i_vs gv4i_t;
typedef __attribute__((address_space(3))) v4i_vs lv4i_t;
#else
#define USE_ASYNC_LDS 0
#endif

__device__ __forceinline__ void cp16_g2l(void* lds, const void* g) {
#if USE_ASYNC_LDS
    __builtin_amdgcn_global_load_async_to_lds_b128(
        (gv4i_t*)g, (lv4i_t*)lds, 0, 0);
#else
    *(v8bf*)lds = *(const v8bf*)g;
#endif
}

__device__ __forceinline__ void wait_async_cp() {
#if USE_ASYNC_LDS
#if __has_builtin(__builtin_amdgcn_s_wait_asynccnt)
    __builtin_amdgcn_s_wait_asynccnt(0);
#else
    asm volatile("s_wait_asynccnt 0" ::: "memory");
#endif
#endif
}

// ---------------------------------------------------------------------------
// fp32 -> bf16 elementwise convert (vectorized x4)
// ---------------------------------------------------------------------------
__global__ void cvt_f32_bf16(const float* __restrict__ in, __bf16* __restrict__ out, int n4) {
    int i = blockIdx.x * blockDim.x + threadIdx.x;
    if (i < n4) {
        float4 v = ((const float4*)in)[i];
        v4bf o;
        o[0] = (__bf16)v.x; o[1] = (__bf16)v.y; o[2] = (__bf16)v.z; o[3] = (__bf16)v.w;
        ((v4bf*)out)[i] = o;
    }
}

// ---------------------------------------------------------------------------
// fp32 [R][C] -> bf16 transpose [C][R] (tiled via LDS)
// ---------------------------------------------------------------------------
__global__ void transpose_cvt(const float* __restrict__ in, __bf16* __restrict__ out,
                              int R, int C) {
    __shared__ float tile[32][33];
    const int tx = threadIdx.x, ty = threadIdx.y;
    const int c0 = blockIdx.x * 32, r0 = blockIdx.y * 32;
#pragma unroll
    for (int i = 0; i < 4; i++) {
        int r = r0 + ty + i * 8;
        tile[ty + i * 8][tx] = in[(size_t)r * C + c0 + tx];
    }
    __syncthreads();
#pragma unroll
    for (int i = 0; i < 4; i++) {
        int orow = c0 + ty + i * 8;   // out row = in col
        int ocol = r0 + tx;           // out col = in row
        out[(size_t)orow * R + ocol] = (__bf16)tile[tx][ty + i * 8];
    }
}

// ---------------------------------------------------------------------------
// Build KV^T [B][D][KD] from fLL [B][NL][D] and fVV [B][NV][D] (bf16 transpose)
// ---------------------------------------------------------------------------
__global__ void build_kvt(const __bf16* __restrict__ fLL, const __bf16* __restrict__ fVV,
                          __bf16* __restrict__ kvt) {
    __shared__ __bf16 t[32][33];
    const int z = blockIdx.z;
    const __bf16* fL = fLL + (size_t)z * NL_ * D_;
    const __bf16* fV = fVV + (size_t)z * NV_ * D_;
    __bf16* o = kvt + (size_t)z * D_ * KD_;
    const int tx = threadIdx.x, ty = threadIdx.y;
    const int k0 = blockIdx.x * 32, d0 = blockIdx.y * 32;
#pragma unroll
    for (int i = 0; i < 4; i++) {
        int k = k0 + ty + i * 8;
        int d = d0 + tx;
        __bf16 v = (k < NL_) ? fL[(size_t)k * D_ + d] : fV[(size_t)(k - NL_) * D_ + d];
        t[ty + i * 8][tx] = v;
    }
    __syncthreads();
#pragma unroll
    for (int i = 0; i < 4; i++) {
        int d = d0 + ty + i * 8;
        int k = k0 + tx;
        o[(size_t)d * KD_ + k] = t[tx][ty + i * 8];
    }
}

// ---------------------------------------------------------------------------
// Softmax over axis=1 (rows of A [B][NV][KD]); one column per thread.
// ---------------------------------------------------------------------------
__global__ void softmax_dim1(const float* __restrict__ Aout, __bf16* __restrict__ attnb) {
    int idx = blockIdx.x * blockDim.x + threadIdx.x;  // 0 .. B*KD-1
    int b = idx / KD_, k = idx % KD_;
    const float* col = Aout + (size_t)b * NV_ * KD_ + k;
    float m = -__builtin_inff();
    for (int n = 0; n < NV_; n++) m = fmaxf(m, col[(size_t)n * KD_]);
    float s = 0.f;
    for (int n = 0; n < NV_; n++) s += __expf(col[(size_t)n * KD_] - m);
    float inv = 1.f / s;
    __bf16* oc = attnb + (size_t)b * NV_ * KD_ + k;
    for (int n = 0; n < NV_; n++)
        oc[(size_t)n * KD_] = (__bf16)(__expf(col[(size_t)n * KD_] - m) * inv);
}

// ---------------------------------------------------------------------------
// Generic NT GEMM: C[m,n] = epilogue( sum_k A[m,k] * B[n,k] )
// A: bf16 [M][K], B: bf16 [N][K], K % 32 == 0, M % 128 == 0, N % 256 == 0.
// MODE 0: bf16 out = relu(acc + bias[n]); MODE 1: bf16 out = acc + bias[n];
// MODE 2: f32 out = acc * scale;          MODE 3: f32 out = acc.
// ---------------------------------------------------------------------------
template <int MODE>
__global__ __launch_bounds__(256)
void gemm_nt_bf16(const __bf16* __restrict__ Ag, const __bf16* __restrict__ Bg,
                  void* __restrict__ Cg, const float* __restrict__ bias,
                  int M, int N, int K, int ldc, float scale,
                  long long sA, long long sB, long long sC) {
    constexpr int BM = 128, BN = 256, BK = 32, PAD = 40;
    __shared__ __align__(16) __bf16 As[2][BM][PAD];
    __shared__ __align__(16) __bf16 Bs[2][BN][PAD];

    const int z = blockIdx.z;
    const __bf16* Ab = Ag + (size_t)z * sA;
    const __bf16* Bb = Bg + (size_t)z * sB;

    const int mBlock = blockIdx.y * BM;
    const int nBlock = blockIdx.x * BN;

    const int tid   = threadIdx.x;
    const int wave  = tid >> 5;
    const int lane  = tid & 31;
    const int hf    = lane >> 4;      // half-wave
    const int lr    = lane & 15;
    const int waveM = wave & 1;       // 0..1  -> 64-row strip
    const int waveN = wave >> 1;      // 0..3  -> 64-col strip

    v8f acc[4][4];
#pragma unroll
    for (int mi = 0; mi < 4; mi++)
#pragma unroll
        for (int ni = 0; ni < 4; ni++)
#pragma unroll
            for (int i = 0; i < 8; i++) acc[mi][ni][i] = 0.f;

    const int ldr = tid >> 2;            // 0..63 : row stripe for staging
    const int kc  = (tid & 3) * 8;       // 8-elem chunk within 32-wide K tile

    // Stage one 128x32 A tile + 256x32 B tile into LDS buffer `buf`.
    auto stage = [&](int buf, int kt) {
#pragma unroll
        for (int r = 0; r < 2; r++) {
            int row = ldr + r * 64;
            cp16_g2l(&As[buf][row][kc], Ab + (size_t)(mBlock + row) * K + kt + kc);
        }
#pragma unroll
        for (int r = 0; r < 4; r++) {
            int row = ldr + r * 64;
            cp16_g2l(&Bs[buf][row][kc], Bb + (size_t)(nBlock + row) * K + kt + kc);
        }
    };

    const int nk = K / BK;
    stage(0, 0);
    wait_async_cp();
    __syncthreads();

    for (int t = 0; t < nk; t++) {
        const int buf = t & 1;
        if (t + 1 < nk) stage(buf ^ 1, (t + 1) * BK);   // prefetch next tile

        // A fragments: lane lr = row; K halves {hf*8..+7, 16+hf*8..+7}
        v16bf af[4];
#pragma unroll
        for (int mi = 0; mi < 4; mi++) {
            const __bf16* p = &As[buf][waveM * 64 + mi * 16 + lr][hf * 8];
            v8bf lo = *(const v8bf*)p;
            v8bf hi = *(const v8bf*)(p + 16);
#pragma unroll
            for (int i = 0; i < 8; i++) { af[mi][i] = lo[i]; af[mi][i + 8] = hi[i]; }
        }
        // B fragments: lane lr = column; K halves {hf*8..+7, 16+hf*8..+7}
        v16bf bfr[4];
#pragma unroll
        for (int ni = 0; ni < 4; ni++) {
            const __bf16* p = &Bs[buf][waveN * 64 + ni * 16 + lr][hf * 8];
            v8bf lo = *(const v8bf*)p;
            v8bf hi = *(const v8bf*)(p + 16);
#pragma unroll
            for (int i = 0; i < 8; i++) { bfr[ni][i] = lo[i]; bfr[ni][i + 8] = hi[i]; }
        }
#pragma unroll
        for (int mi = 0; mi < 4; mi++)
#pragma unroll
            for (int ni = 0; ni < 4; ni++)
                acc[mi][ni] = __builtin_amdgcn_wmma_f32_16x16x32_bf16(
                    false, af[mi], false, bfr[ni], (short)0, acc[mi][ni], false, false);

        wait_async_cp();   // prefetch into buf^1 complete
        __syncthreads();   // all waves done reading buf; LDS writes visible
    }

    // Epilogue. C layout: VGPR i -> row (i + hf*8), col = lr (per 16x16 tile).
#pragma unroll
    for (int mi = 0; mi < 4; mi++) {
        int gm0 = mBlock + waveM * 64 + mi * 16 + hf * 8;
#pragma unroll
        for (int ni = 0; ni < 4; ni++) {
            int gn = nBlock + waveN * 64 + ni * 16 + lr;
#pragma unroll
            for (int i = 0; i < 8; i++) {
                float v = acc[mi][ni][i];
                size_t off = (size_t)(gm0 + i) * ldc + gn;
                if constexpr (MODE == 0) {
                    (((__bf16*)Cg) + (size_t)z * sC)[off] = (__bf16)fmaxf(v + bias[gn], 0.f);
                } else if constexpr (MODE == 1) {
                    (((__bf16*)Cg) + (size_t)z * sC)[off] = (__bf16)(v + bias[gn]);
                } else if constexpr (MODE == 2) {
                    (((float*)Cg) + (size_t)z * sC)[off] = v * scale;
                } else {
                    (((float*)Cg) + (size_t)z * sC)[off] = v;
                }
            }
        }
    }
}

// ---------------------------------------------------------------------------
// Host orchestration
// ---------------------------------------------------------------------------
extern "C" void kernel_launch(void* const* d_in, const int* in_sizes, int n_in,
                              void* d_out, int out_size, void* d_ws, size_t ws_size,
                              hipStream_t stream) {
    (void)in_sizes; (void)n_in; (void)out_size; (void)ws_size;

    const float* V    = (const float*)d_in[0];
    const float* L    = (const float*)d_in[1];
    // d_in[2], d_in[3] = masks (unused by the reference forward)
    const float* w1_V = (const float*)d_in[4];
    const float* b1_V = (const float*)d_in[5];
    const float* w2_V = (const float*)d_in[6];
    const float* b2_V = (const float*)d_in[7];
    const float* w1_L = (const float*)d_in[8];
    const float* b1_L = (const float*)d_in[9];
    const float* w2_L = (const float*)d_in[10];
    const float* b2_L = (const float*)d_in[11];

    float* outA = (float*)d_out;                                   // [B][NV][KD]
    float* outV = (float*)d_out + (size_t)Bb_ * NV_ * KD_;         // [B][NV][D]

    char* ws = (char*)d_ws;
    size_t off = 0;
    auto alloc = [&](size_t elems) { char* p = ws + off; off += elems * sizeof(__bf16); return (__bf16*)p; };
    __bf16* Vb    = alloc((size_t)MV_ * D_);        // bf16 V
    __bf16* Lb    = alloc((size_t)ML_ * D_);        // bf16 L
    __bf16* W1VT  = alloc((size_t)H_ * D_);         // w1_V^T  [H][D]
    __bf16* W1LT  = alloc((size_t)H_ * D_);         // w1_L^T  [H][D]
    __bf16* W2VT  = alloc((size_t)D_ * H_);         // w2_V^T  [D][H]
    __bf16* W2LT  = alloc((size_t)D_ * H_);         // w2_L^T  [D][H]
    __bf16* hid   = alloc((size_t)MV_ * H_);        // hidden (reused by all 3 FFNs)
    __bf16* fVVb  = alloc((size_t)MV_ * D_);        // ffn_V(V)
    __bf16* fLLb  = alloc((size_t)ML_ * D_);        // ffn_L(L)
    __bf16* fLVb  = alloc((size_t)MV_ * D_);        // ffn_L(V)
    __bf16* attnb = alloc((size_t)Bb_ * NV_ * KD_); // softmax(A, axis=1) in bf16
    __bf16* KVT   = alloc((size_t)Bb_ * D_ * KD_);  // KV^T per batch

    const float scale = 1.0f / 32.0f;               // 1/sqrt(D)
    dim3 blk256(256), blkT(32, 8);

    // 1) convert activations to bf16
    cvt_f32_bf16<<<(MV_ * D_ / 4 + 255) / 256, blk256, 0, stream>>>(V, Vb, MV_ * D_ / 4);
    cvt_f32_bf16<<<(ML_ * D_ / 4 + 255) / 256, blk256, 0, stream>>>(L, Lb, ML_ * D_ / 4);

    // 2) transpose+convert weights: w1 [D][H] -> [H][D]; w2 [H][D] -> [D][H]
    transpose_cvt<<<dim3(H_ / 32, D_ / 32), blkT, 0, stream>>>(w1_V, W1VT, D_, H_);
    transpose_cvt<<<dim3(H_ / 32, D_ / 32), blkT, 0, stream>>>(w1_L, W1LT, D_, H_);
    transpose_cvt<<<dim3(D_ / 32, H_ / 32), blkT, 0, stream>>>(w2_V, W2VT, H_, D_);
    transpose_cvt<<<dim3(D_ / 32, H_ / 32), blkT, 0, stream>>>(w2_L, W2LT, H_, D_);

    // 3) fV_V = ffn_V(V)
    gemm_nt_bf16<0><<<dim3(H_ / 256, MV_ / 128, 1), blk256, 0, stream>>>(
        Vb, W1VT, hid, b1_V, MV_, H_, D_, H_, 1.f, 0, 0, 0);
    gemm_nt_bf16<1><<<dim3(D_ / 256, MV_ / 128, 1), blk256, 0, stream>>>(
        hid, W2VT, fVVb, b2_V, MV_, D_, H_, D_, 1.f, 0, 0, 0);

    // 4) fL_L = ffn_L(L)
    gemm_nt_bf16<0><<<dim3(H_ / 256, ML_ / 128, 1), blk256, 0, stream>>>(
        Lb, W1LT, hid, b1_L, ML_, H_, D_, H_, 1.f, 0, 0, 0);
    gemm_nt_bf16<1><<<dim3(D_ / 256, ML_ / 128, 1), blk256, 0, stream>>>(
        hid, W2LT, fLLb, b2_L, ML_, D_, H_, D_, 1.f, 0, 0, 0);

    // 5) fL_V = ffn_L(V)
    gemm_nt_bf16<0><<<dim3(H_ / 256, MV_ / 128, 1), blk256, 0, stream>>>(
        Vb, W1LT, hid, b1_L, MV_, H_, D_, H_, 1.f, 0, 0, 0);
    gemm_nt_bf16<1><<<dim3(D_ / 256, MV_ / 128, 1), blk256, 0, stream>>>(
        hid, W2LT, fLVb, b2_L, MV_, D_, H_, D_, 1.f, 0, 0, 0);

    // 6) A_LV = scale * fV_V @ fL_L^T  -> d_out cols [0, NL)
    gemm_nt_bf16<2><<<dim3(NL_ / 256, NV_ / 128, Bb_), blk256, 0, stream>>>(
        fVVb, fLLb, outA, nullptr, NV_, NL_, D_, KD_, scale,
        (long long)NV_ * D_, (long long)NL_ * D_, (long long)NV_ * KD_);

    // 7) A_VV = scale * fV_V @ fL_V^T  -> d_out cols [NL, NL+NV)
    gemm_nt_bf16<2><<<dim3(NV_ / 256, NV_ / 128, Bb_), blk256, 0, stream>>>(
        fVVb, fLVb, outA + NL_, nullptr, NV_, NV_, D_, KD_, scale,
        (long long)NV_ * D_, (long long)NV_ * D_, (long long)NV_ * KD_);

    // 8) attn = softmax(A, axis=1) -> bf16
    softmax_dim1<<<(Bb_ * KD_) / 256, blk256, 0, stream>>>(outA, attnb);

    // 9) KV^T build
    build_kvt<<<dim3(KD_ / 32, D_ / 32, Bb_), blkT, 0, stream>>>(fLLb, fVVb, KVT);

    // 10) V_next = attn @ KV  (NT with KV^T)
    gemm_nt_bf16<3><<<dim3(D_ / 256, NV_ / 128, Bb_), blk256, 0, stream>>>(
        attnb, KVT, outV, nullptr, NV_, D_, KD_, D_, 1.f,
        (long long)NV_ * KD_, (long long)D_ * KD_, (long long)NV_ * D_);
}
